// InvResMLPBlock_33079838113814
// MI455X (gfx1250) — matrix operand order repaired and checked by reading, not compiled, not run on hardware
//
#include <hip/hip_runtime.h>
#include <hip/hip_bf16.h>

typedef __attribute__((ext_vector_type(16))) __bf16 v16bf;
typedef __attribute__((ext_vector_type(8)))  __bf16 v8bf;
typedef __attribute__((ext_vector_type(8)))  float  v8f;

#define B_    4
#define N_    4096
#define C_    128
#define H_    512
#define S_    32
#define KPAD  160          // (C_+3)=131 padded to 5 * 32 (bf16 WMMA K)
#define RAD   0.1f
#define EPS_  1e-5f

__device__ __forceinline__ v16bf cat16(v8bf lo, v8bf hi) {
  return __builtin_shufflevector(lo, hi, 0,1,2,3,4,5,6,7,8,9,10,11,12,13,14,15);
}

// ---------------------------------------------------------------------------
// Kernel 0: convert weights to bf16 (N-major rows, contiguous K) + fold BN.
// ---------------------------------------------------------------------------
__global__ void prep_kernel(const float* __restrict__ Wn, const float* __restrict__ W1,
                            const float* __restrict__ W2,
                            const float* gn, const float* bnb, const float* mn, const float* vn,
                            const float* g1, const float* b1, const float* m1, const float* v1,
                            const float* g2, const float* b2, const float* m2, const float* v2,
                            __bf16* wn_bf, __bf16* w1_bf, __bf16* w2_bf,
                            float* sA0, float* sB0, float* sA1, float* sB1,
                            float* sA2, float* sB2)
{
  int i = blockIdx.x * blockDim.x + threadIdx.x;
  if (i < C_ * KPAD) {
    int o = i / KPAD, c = i - o * KPAD;
    wn_bf[i] = (c < C_ + 3) ? (__bf16)Wn[o * (C_ + 3) + c] : (__bf16)0.0f;
  }
  if (i < H_ * C_) w1_bf[i] = (__bf16)W1[i];
  if (i < C_ * H_) w2_bf[i] = (__bf16)W2[i];
  if (i < C_) { float s = gn[i] * rsqrtf(vn[i] + EPS_); sA0[i] = s; sB0[i] = bnb[i] - mn[i] * s; }
  if (i < H_) { float s = g1[i] * rsqrtf(v1[i] + EPS_); sA1[i] = s; sB1[i] = b1[i] - m1[i] * s; }
  if (i < C_) { float s = g2[i] * rsqrtf(v2[i] + EPS_); sA2[i] = s; sB2[i] = b2[i] - m2[i] * s; }
}

// ---------------------------------------------------------------------------
// Kernel 1: ball query. One wave per point; ordered compaction of the first
// 32 in-radius indices via ballot/popc, pad remainder with first hit.
// ---------------------------------------------------------------------------
__global__ void __launch_bounds__(256)
ballquery_kernel(const float* __restrict__ xyz, int* __restrict__ idxbuf)
{
  const int tid  = threadIdx.x;
  const int lane = tid & 31;
  const int p    = blockIdx.x * 8 + (tid >> 5);
  const int b    = p >> 12;                 // N_ = 4096
  const int n    = p & (N_ - 1);
  const float* xb = xyz + (size_t)b * N_ * 3;
  const float cx = xb[n * 3 + 0], cy = xb[n * 3 + 1], cz = xb[n * 3 + 2];
  const float r2 = RAD * RAD;
  int* out = idxbuf + (size_t)p * S_;

  int count = 0;
  int first = n;
  for (int j0 = 0; j0 < N_ && count < S_; j0 += 32) {
    int j = j0 + lane;
    float dx = xb[j * 3 + 0] - cx;
    float dy = xb[j * 3 + 1] - cy;
    float dz = xb[j * 3 + 2] - cz;
    float d2 = dx * dx + dy * dy + dz * dz;
    unsigned mask = (unsigned)__ballot(d2 <= r2);
    if (count == 0 && mask) first = j0 + (__ffs(mask) - 1);
    int pre = __popc(mask & ((1u << lane) - 1u));
    if ((mask >> lane) & 1u) {
      int slot = count + pre;
      if (slot < S_) out[slot] = j;
    }
    count += __popc(mask);
  }
  if (count > S_) count = S_;
  if (lane >= count) out[lane] = first;     // pad with first neighbor
}

// ---------------------------------------------------------------------------
// Kernel 2: one block per point. Stage grouped (32 x 160) bf16 in LDS,
// GEMM (M=32 samples, N=128 out-ch, K=160) with v_wmma_f32_16x16x32_bf16,
// fused BN+ReLU+max over samples -> x[p][128].
// Wave w owns n-tile w: 2 m-tiles x 5 k-steps = 10 WMMAs.
// ---------------------------------------------------------------------------
__global__ void __launch_bounds__(256)
group_conv_max_kernel(const float* __restrict__ xyz, const float* __restrict__ feats,
                      const int* __restrict__ idxbuf, const __bf16* __restrict__ wn_bf,
                      const float* __restrict__ scl, const float* __restrict__ sh,
                      float* __restrict__ xout)
{
  __shared__ int   sIdx[S_];
  __shared__ float sCen[3];
  __shared__ __attribute__((aligned(16))) __bf16 sA[S_ * KPAD];

  const int tid = threadIdx.x;
  const int p   = blockIdx.x;
  const int b   = p >> 12;
  const int n   = p & (N_ - 1);

  if (tid < S_) sIdx[tid] = idxbuf[(size_t)p * S_ + tid];
  if (tid < 3)  sCen[tid] = xyz[((size_t)b * N_ + n) * 3 + tid];
  __syncthreads();

  for (int e = tid; e < S_ * KPAD; e += 256) {
    int s = e / KPAD, c = e - s * KPAD;
    int j = sIdx[s];
    float v = 0.0f;
    if (c < 3)            v = (xyz[((size_t)b * N_ + j) * 3 + c] - sCen[c]) * (1.0f / RAD);
    else if (c < C_ + 3)  v = feats[((size_t)b * N_ + j) * C_ + (c - 3)];
    sA[e] = (__bf16)v;
  }
  __syncthreads();

  const int wave = tid >> 5, lane = tid & 31;
  const int half = lane >> 4, l16 = lane & 15;
  const int col  = wave * 16 + l16;                 // output channel o

  v8f acc0 = {}; v8f acc1 = {};
  const __bf16* brow = wn_bf + (size_t)col * KPAD;
  #pragma unroll
  for (int k = 0; k < 5; ++k) {
    const int kb = k * 32;
    v16bf bf = cat16(*(const v8bf*)(brow + kb + half * 8),
                     *(const v8bf*)(brow + kb + 16 + half * 8));
    {
      const __bf16* arow = sA + l16 * KPAD + kb;
      v16bf af = cat16(*(const v8bf*)(arow + half * 8),
                       *(const v8bf*)(arow + 16 + half * 8));
      acc0 = __builtin_amdgcn_wmma_f32_16x16x32_bf16(false, af, false, bf,
                                                     (short)0, acc0, false, false);
    }
    {
      const __bf16* arow = sA + (16 + l16) * KPAD + kb;
      v16bf af = cat16(*(const v8bf*)(arow + half * 8),
                       *(const v8bf*)(arow + 16 + half * 8));
      acc1 = __builtin_amdgcn_wmma_f32_16x16x32_bf16(false, af, false, bf,
                                                     (short)0, acc1, false, false);
    }
  }
  const float scale = scl[col], bias = sh[col];
  float mx = 0.0f;                                  // ReLU => max >= 0
  #pragma unroll
  for (int r = 0; r < 8; ++r) {
    float t0 = fmaxf(acc0[r] * scale + bias, 0.0f);
    float t1 = fmaxf(acc1[r] * scale + bias, 0.0f);
    mx = fmaxf(mx, fmaxf(t0, t1));
  }
  mx = fmaxf(mx, __shfl_xor(mx, 16, 32));
  if (lane < 16) xout[(size_t)p * C_ + col] = mx;
}

// ---------------------------------------------------------------------------
// Kernel 3: one block per 16 points. GEMM1 (16x512x128) + BN/ReLU -> LDS,
// GEMM2 (16x128x512) + BN + residual + ReLU -> out. All via bf16 WMMA.
// ---------------------------------------------------------------------------
__global__ void __launch_bounds__(256)
mlp_kernel(const float* __restrict__ xbuf, const float* __restrict__ feats,
           const __bf16* __restrict__ w1_bf, const __bf16* __restrict__ w2_bf,
           const float* __restrict__ s1v, const float* __restrict__ h1v,
           const float* __restrict__ s2v, const float* __restrict__ h2v,
           float* __restrict__ out)
{
  __shared__ __attribute__((aligned(16))) __bf16 sX[16 * C_];
  __shared__ __attribute__((aligned(16))) __bf16 sH[16 * H_];

  const int tid = threadIdx.x;
  const int pt0 = blockIdx.x * 16;                  // global point base (incl. batch)

  for (int e = tid; e < 16 * C_; e += 256)
    sX[e] = (__bf16)xbuf[(size_t)pt0 * C_ + e];
  __syncthreads();

  const int wave = tid >> 5, lane = tid & 31;
  const int half = lane >> 4, l16 = lane & 15;

  // GEMM1: M=16 points, N=512 hidden, K=128. Each wave: 4 n-tiles x 4 k-steps.
  #pragma unroll
  for (int nt = 0; nt < 4; ++nt) {
    const int col = (wave * 4 + nt) * 16 + l16;     // hidden channel
    v8f acc = {};
    const __bf16* brow = w1_bf + (size_t)col * C_;
    #pragma unroll
    for (int k = 0; k < 4; ++k) {
      const int kb = k * 32;
      v16bf bf = cat16(*(const v8bf*)(brow + kb + half * 8),
                       *(const v8bf*)(brow + kb + 16 + half * 8));
      const __bf16* arow = sX + l16 * C_ + kb;
      v16bf af = cat16(*(const v8bf*)(arow + half * 8),
                       *(const v8bf*)(arow + 16 + half * 8));
      acc = __builtin_amdgcn_wmma_f32_16x16x32_bf16(false, af, false, bf,
                                                    (short)0, acc, false, false);
    }
    const float scale = s1v[col], bias = h1v[col];
    #pragma unroll
    for (int r = 0; r < 8; ++r) {
      int m = r + half * 8;
      sH[m * H_ + col] = (__bf16)fmaxf(acc[r] * scale + bias, 0.0f);
    }
  }
  __syncthreads();

  // GEMM2: M=16 points, N=128 out-ch, K=512. Each wave: 1 n-tile x 16 k-steps.
  {
    const int col = wave * 16 + l16;                // output channel
    v8f acc = {};
    const __bf16* brow = w2_bf + (size_t)col * H_;
    #pragma unroll
    for (int k = 0; k < 16; ++k) {
      const int kb = k * 32;
      v16bf bf = cat16(*(const v8bf*)(brow + kb + half * 8),
                       *(const v8bf*)(brow + kb + 16 + half * 8));
      const __bf16* arow = sH + l16 * H_ + kb;
      v16bf af = cat16(*(const v8bf*)(arow + half * 8),
                       *(const v8bf*)(arow + 16 + half * 8));
      acc = __builtin_amdgcn_wmma_f32_16x16x32_bf16(false, af, false, bf,
                                                    (short)0, acc, false, false);
    }
    const float scale = s2v[col], bias = h2v[col];
    #pragma unroll
    for (int r = 0; r < 8; ++r) {
      int m = r + half * 8;
      size_t off = ((size_t)(pt0 + m)) * C_ + col;
      out[off] = fmaxf(acc[r] * scale + bias + feats[off], 0.0f);
    }
  }
}

// ---------------------------------------------------------------------------
extern "C" void kernel_launch(void* const* d_in, const int* in_sizes, int n_in,
                              void* d_out, int out_size, void* d_ws, size_t ws_size,
                              hipStream_t stream)
{
  const float* xyz  = (const float*)d_in[0];
  const float* feats= (const float*)d_in[1];
  const float* Wn   = (const float*)d_in[2];
  const float* gn   = (const float*)d_in[3];
  const float* bnb  = (const float*)d_in[4];
  const float* mn   = (const float*)d_in[5];
  const float* vn   = (const float*)d_in[6];
  const float* W1   = (const float*)d_in[7];
  const float* g1   = (const float*)d_in[8];
  const float* b1   = (const float*)d_in[9];
  const float* m1   = (const float*)d_in[10];
  const float* v1   = (const float*)d_in[11];
  const float* W2   = (const float*)d_in[12];
  const float* g2   = (const float*)d_in[13];
  const float* b2   = (const float*)d_in[14];
  const float* m2   = (const float*)d_in[15];
  const float* v2   = (const float*)d_in[16];
  float* out = (float*)d_out;

  char* ws = (char*)d_ws;
  size_t off = 0;
  auto alloc = [&](size_t bytes) -> void* {
    void* pp = ws + off;
    off = (off + bytes + 255) & ~(size_t)255;
    return pp;
  };

  int*    idxbuf = (int*)   alloc((size_t)B_ * N_ * S_ * sizeof(int));
  float*  xbuf   = (float*) alloc((size_t)B_ * N_ * C_ * sizeof(float));
  __bf16* wn_bf  = (__bf16*)alloc((size_t)C_ * KPAD * 2);
  __bf16* w1_bf  = (__bf16*)alloc((size_t)H_ * C_ * 2);
  __bf16* w2_bf  = (__bf16*)alloc((size_t)C_ * H_ * 2);
  float*  sA0 = (float*)alloc(C_ * 4); float* sB0 = (float*)alloc(C_ * 4);
  float*  sA1 = (float*)alloc(H_ * 4); float* sB1 = (float*)alloc(H_ * 4);
  float*  sA2 = (float*)alloc(C_ * 4); float* sB2 = (float*)alloc(C_ * 4);

  prep_kernel<<<(H_ * C_ + 255) / 256, 256, 0, stream>>>(
      Wn, W1, W2, gn, bnb, mn, vn, g1, b1, m1, v1, g2, b2, m2, v2,
      wn_bf, w1_bf, w2_bf, sA0, sB0, sA1, sB1, sA2, sB2);

  ballquery_kernel<<<B_ * N_ / 8, 256, 0, stream>>>(xyz, idxbuf);

  group_conv_max_kernel<<<B_ * N_, 256, 0, stream>>>(
      xyz, feats, idxbuf, wn_bf, sA0, sB0, xbuf);

  mlp_kernel<<<B_ * N_ / 16, 256, 0, stream>>>(
      xbuf, feats, w1_bf, w2_bf, sA1, sB1, sA2, sB2, out);
}
